// CNN_OTAM_91010357002277
// MI455X (gfx1250) — matrix operand order repaired and checked by reading, not compiled, not run on hardware
//
#include <hip/hip_runtime.h>
#include <stdint.h>

#define LBDA 0.1f

typedef __attribute__((ext_vector_type(16))) __bf16        bf16x16;
typedef __attribute__((ext_vector_type(8)))  float         f32x8;
typedef __attribute__((ext_vector_type(4)))  unsigned int  u32x4;

union ABFrag { u32x4 u[2]; bf16x16 v; };

__device__ __forceinline__ unsigned short f32_to_bf16_rne(float f) {
    union { float f; unsigned int u; } c; c.f = f;
    unsigned int x = c.u;
    unsigned int r = x + 0x7FFFu + ((x >> 16) & 1u);   // round-to-nearest-even
    return (unsigned short)(r >> 16);
}

// ---------------------------------------------------------------------------
// Kernel 1: fp32 -> bf16 conversion + L2 row norms. One wave32 per 2048-row.
// ---------------------------------------------------------------------------
__global__ __launch_bounds__(256)
void convert_norm_kernel(const float* __restrict__ src,
                         unsigned short* __restrict__ dst,
                         float* __restrict__ norms,
                         int nrows) {
    const int wave = (int)((blockIdx.x * blockDim.x + threadIdx.x) >> 5);
    const int lane = (int)(threadIdx.x & 31);
    if (wave >= nrows) return;
    const float4* srow = (const float4*)(src + (size_t)wave * 2048);
    uint2*        drow = (uint2*)(dst + (size_t)wave * 2048);
    float ss = 0.f;
    #pragma unroll
    for (int j = 0; j < 16; ++j) {
        const int i4 = lane + j * 32;               // 512 float4 per row
        const float4 v = srow[i4];
        ss += v.x * v.x + v.y * v.y + v.z * v.z + v.w * v.w;
        uint2 p;
        p.x = (unsigned)f32_to_bf16_rne(v.x) | ((unsigned)f32_to_bf16_rne(v.y) << 16);
        p.y = (unsigned)f32_to_bf16_rne(v.z) | ((unsigned)f32_to_bf16_rne(v.w) << 16);
        drow[i4] = p;
    }
    #pragma unroll
    for (int off = 16; off >= 1; off >>= 1)
        ss += __shfl_xor(ss, off, 32);
    if (lane == 0) norms[wave] = sqrtf(ss);
}

// ---------------------------------------------------------------------------
// Kernel 2: bf16 WMMA GEMM (M=8192, N=400, K=2048) fused with cosine-distance
// epilogue. One wave per 2 M-tiles x 5 N-tiles (10 accumulators); 1280 waves.
// A = target (8192 x 2048, K-contig), B = support (400 x 2048, K-contig = B^T).
// ---------------------------------------------------------------------------
__global__ __launch_bounds__(256)
void gemm_dist_kernel(const unsigned short* __restrict__ A,
                      const unsigned short* __restrict__ B,
                      const float* __restrict__ qn,   // 8192 target row norms
                      const float* __restrict__ sn,   // 400 support row norms
                      float* __restrict__ D) {        // 8192 x 400 distances
    const int wave = (int)((blockIdx.x * blockDim.x + threadIdx.x) >> 5);
    const int lane = (int)(threadIdx.x & 31);
    const int mpair = wave / 5;          // 0..255 -> M-tiles 2*mpair, 2*mpair+1
    const int ngrp  = wave % 5;          // 5 N-tiles each
    const int hf    = lane >> 4;         // lane half (0/1)
    const int l15   = lane & 15;

    // A rows for this lane, for the two M-tiles
    size_t arow[2];
    #pragma unroll
    for (int t = 0; t < 2; ++t)
        arow[t] = (size_t)((mpair * 2 + t) * 16 + l15) * 2048;

    const f32x8 zero = {};
    f32x8 acc[2][5];
    #pragma unroll
    for (int t = 0; t < 2; ++t)
        #pragma unroll
        for (int i = 0; i < 5; ++i) acc[t][i] = zero;

    size_t brow[5];
    #pragma unroll
    for (int i = 0; i < 5; ++i)
        brow[i] = (size_t)((ngrp * 5 + i) * 16 + l15) * 2048;

    for (int kk = 0; kk < 2048; kk += 32) {
        // A 16x32 bf16 fragments: lane l15 = row M; halves hold
        // K = 8*hf..8*hf+7 and 16+8*hf..16+8*hf+7
        ABFrag a[2];
        #pragma unroll
        for (int t = 0; t < 2; ++t) {
            a[t].u[0] = *(const u32x4*)(A + arow[t] + kk + hf * 8);
            a[t].u[1] = *(const u32x4*)(A + arow[t] + kk + 16 + hf * 8);
        }
        const int k0 = kk + hf * 8;      // B: lane holds K = 8*hf..+7 and 16+8*hf..+7
        #pragma unroll
        for (int i = 0; i < 5; ++i) {
            ABFrag b;
            b.u[0] = *(const u32x4*)(B + brow[i] + k0);
            b.u[1] = *(const u32x4*)(B + brow[i] + k0 + 16);
            #pragma unroll
            for (int t = 0; t < 2; ++t)
                acc[t][i] = __builtin_amdgcn_wmma_f32_16x16x32_bf16(
                    false, a[t].v, false, b.v, (short)0, acc[t][i], false, false);
        }
    }

    // Epilogue: dist = 1 - dot / (|q| * |s|).  C layout: lane half picks M rows
    // v + 8*hf, column N = l15.
    #pragma unroll
    for (int i = 0; i < 5; ++i) {
        const int n = (ngrp * 5 + i) * 16 + l15;
        const float snv = sn[n];
        #pragma unroll
        for (int t = 0; t < 2; ++t) {
            #pragma unroll
            for (int v = 0; v < 8; ++v) {
                const int mm = (mpair * 2 + t) * 16 + v + 8 * hf;
                const float dot = acc[t][i][v];
                D[(size_t)mm * 400 + n] = 1.0f - dot / (qn[mm] * snv);
            }
        }
    }
}

// ---------------------------------------------------------------------------
// Kernel 3: OTAM soft-DTW DP, one lane per (pair, direction) problem.
// D layout: element (q,k,t,s) at D[(q*16 + t)*400 + k*16 + s].
// ---------------------------------------------------------------------------
__global__ __launch_bounds__(256)
void otam_dp_kernel(const float* __restrict__ D,
                    float* __restrict__ abuf,
                    float* __restrict__ bbuf) {
    const int tid = (int)(blockIdx.x * blockDim.x + threadIdx.x);
    if (tid >= 25600) return;
    const int dir  = tid >= 12800;
    const int pair = dir ? tid - 12800 : tid;
    const int q = pair / 25, k = pair % 25;
    const float* base = D + (size_t)q * 6400 + k * 16;  // (t,s) -> base[t*400+s]

    float prev[19];
    #pragma unroll
    for (int j = 0; j < 19; ++j) prev[j] = 0.f;

    for (int r = 0; r < 16; ++r) {
        float cur[19];
        cur[0] = 0.f;
        float left = 0.f;
        #pragma unroll
        for (int j = 0; j < 18; ++j) {
            float cj = 0.f;
            if (j >= 1 && j <= 16) {
                const int c = j - 1;
                cj = dir ? base[c * 400 + r] : base[r * 400 + c];
            }
            const float dup = (j == 1 || j == 17) ? 1.f : 0.f;
            const float dg  = prev[j];
            const float m0  = fminf(left, dg);
            const float s   = __expf((m0 - left) * (1.0f / LBDA)) * (1.f + dup)
                            + __expf((m0 - dg)   * (1.0f / LBDA));
            const float nv  = cj + m0 - LBDA * __logf(s);
            cur[j + 1] = nv;
            left = nv;
        }
        #pragma unroll
        for (int j = 0; j < 19; ++j) prev[j] = cur[j];
    }
    (dir ? bbuf : abuf)[q * 25 + k] = prev[18];
}

// ---------------------------------------------------------------------------
// Kernel 4: class-averaged reduction -> (512, 5)
// ---------------------------------------------------------------------------
__global__ __launch_bounds__(256)
void reduce_kernel(const float* __restrict__ abuf,
                   const float* __restrict__ bbuf,
                   const int* __restrict__ labels,
                   float* __restrict__ out) {
    const int tid = (int)(blockIdx.x * blockDim.x + threadIdx.x);
    if (tid >= 512 * 5) return;
    const int q = tid / 5, c = tid % 5;
    float s = 0.f; int cnt = 0;
    for (int k = 0; k < 25; ++k) {
        if (labels[k] == c) { s += abuf[q * 25 + k] + bbuf[q * 25 + k]; ++cnt; }
    }
    out[tid] = s / (float)cnt;
}

// ---------------------------------------------------------------------------
extern "C" void kernel_launch(void* const* d_in, const int* in_sizes, int n_in,
                              void* d_out, int out_size, void* d_ws, size_t ws_size,
                              hipStream_t stream) {
    const float* sup    = (const float*)d_in[0];   // (25, 16, 2048)
    const float* tgt    = (const float*)d_in[1];   // (512, 16, 2048)
    const int*   labels = (const int*)d_in[2];     // (25,)
    float* out = (float*)d_out;                    // (512, 5)

    char* ws = (char*)d_ws;
    size_t off = 0;
    auto carve = [&](size_t bytes) -> void* {
        void* p = ws + off;
        off += (bytes + 255) & ~(size_t)255;
        return p;
    };
    unsigned short* tgt_bf = (unsigned short*)carve((size_t)8192 * 2048 * 2);
    unsigned short* sup_bf = (unsigned short*)carve((size_t)400 * 2048 * 2);
    float* qn   = (float*)carve(8192 * sizeof(float));
    float* sn   = (float*)carve(400 * sizeof(float));
    float* Dqs  = (float*)carve((size_t)8192 * 400 * sizeof(float));
    float* abuf = (float*)carve(12800 * sizeof(float));
    float* bbuf = (float*)carve(12800 * sizeof(float));

    // 1) convert + norms (8 waves/block, one wave per row)
    convert_norm_kernel<<<1024, 256, 0, stream>>>(tgt, tgt_bf, qn, 8192);
    convert_norm_kernel<<<50,   256, 0, stream>>>(sup, sup_bf, sn, 400);

    // 2) WMMA GEMM + cosine-distance epilogue: 1280 waves = 160 blocks of 8
    gemm_dist_kernel<<<160, 256, 0, stream>>>(tgt_bf, sup_bf, qn, sn, Dqs);

    // 3) 25600 DP problems, one lane each
    otam_dp_kernel<<<100, 256, 0, stream>>>(Dqs, abuf, bbuf);

    // 4) class reduction
    reduce_kernel<<<10, 256, 0, stream>>>(abuf, bbuf, labels, out);
}
